// GenesisLM_49065706389524
// MI455X (gfx1250) — matrix operand order repaired and checked by reading, not compiled, not run on hardware
//
#include <hip/hip_runtime.h>
#include <hip/hip_bf16.h>
#include <math.h>

// ---------------- problem constants (from reference) ----------------
#define BB 2
#define SS 512
#define DD 1024
#define NN 4096
#define LL 4
#define VV 32000
#define TT (BB * SS)     // 1024 tokens
#define D3 (3 * DD)      // 3072

// ---------------- workspace layout (bytes, all 1KB-aligned) ----------------
static constexpr size_t OFF_SCALES = 0;                                   // 16 floats
static constexpr size_t OFF_PART   = 1024;                                // 256 floats
static constexpr size_t OFF_X      = 4096;                                // T*D f32 residual stream
static constexpr size_t OFF_XBF    = OFF_X    + 4ull * TT * DD;           // T*D bf16
static constexpr size_t OFF_GI     = OFF_XBF  + 2ull * TT * DD;           // T*3D f32
static constexpr size_t OFF_HA     = OFF_GI   + 4ull * TT * D3;           // B*D f32
static constexpr size_t OFF_HB     = OFF_HA   + 4ull * BB * DD;           // B*D f32
static constexpr size_t OFF_HID    = OFF_HB   + 4ull * BB * DD;           // T*N bf16
static constexpr size_t OFF_PRELN  = OFF_HID  + 2ull * TT * NN;           // T*D f32

// ---------------- types ----------------
typedef __bf16          v16bf __attribute__((ext_vector_type(16)));
typedef float           v8f   __attribute__((ext_vector_type(8)));
typedef unsigned short  us8   __attribute__((ext_vector_type(8)));
typedef unsigned short  us4   __attribute__((ext_vector_type(4)));
typedef float           f4    __attribute__((ext_vector_type(4)));

union FragU { us8 h[2]; unsigned short us[16]; v16bf v; };   // 32 bytes

__device__ __forceinline__ unsigned short f2bf(float f) {
    unsigned int u = __float_as_uint(f);
    return (unsigned short)((u + 0x7FFFu + ((u >> 16) & 1u)) >> 16);  // RNE
}

__device__ __forceinline__ float sigmoidf_(float x) { return 1.f / (1.f + expf(-x)); }

// ---------------- deterministic abs-mean scale reduction ----------------
__global__ __launch_bounds__(256) void absmean_part_k(const float* __restrict__ w,
                                                      long n, float* __restrict__ part) {
    __shared__ float sm[256];
    float acc = 0.f;
    for (long j = (long)blockIdx.x * 256 + threadIdx.x; j < n; j += 256L * 256L)
        acc += fabsf(w[j]);
    sm[threadIdx.x] = acc;
    __syncthreads();
    for (int s = 128; s > 0; s >>= 1) {
        if ((int)threadIdx.x < s) sm[threadIdx.x] += sm[threadIdx.x + s];
        __syncthreads();
    }
    if (threadIdx.x == 0) part[blockIdx.x] = sm[0];
}

__global__ __launch_bounds__(256) void absmean_fin_k(const float* __restrict__ part,
                                                     float* __restrict__ scales,
                                                     int idx, float inv_n) {
    __shared__ float sm[256];
    sm[threadIdx.x] = part[threadIdx.x];
    __syncthreads();
    for (int s = 128; s > 0; s >>= 1) {
        if ((int)threadIdx.x < s) sm[threadIdx.x] += sm[threadIdx.x + s];
        __syncthreads();
    }
    if (threadIdx.x == 0) scales[idx] = sm[0] * inv_n + 1e-8f;
}

// ---------------- embedding: ternary lookup + pos add ----------------
__global__ __launch_bounds__(256) void embed_k(const int* __restrict__ ids,
                                               const float* __restrict__ emb,
                                               const float* __restrict__ pos,
                                               const float* __restrict__ scales,
                                               float* __restrict__ x,
                                               unsigned short* __restrict__ xbf) {
    long i = (long)blockIdx.x * 256 + threadIdx.x;   // T*D = 1M
    int tok = (int)(i >> 10);
    int d   = (int)(i & (DD - 1));
    int id  = ids[tok];
    float s = scales[0];
    float w = emb[(size_t)id * DD + d];
    float r = fminf(1.f, fmaxf(-1.f, w / s));
    float val = s * rintf(r) + pos[(size_t)(tok & (SS - 1)) * DD + d];
    x[i] = val;
    xbf[i] = f2bf(val);
}

// ---------------- WMMA GEMM with LDS staging + register double-buffering ----------------
// C[M,Nout] = A_bf16[M,K] * q(W[Nout,K])^T, 32x64 output tile per 256-thread block.
// Pipeline per K-tile: ds_store (regs->LDS) | barrier | issue NEXT tile's global loads
// | fragment ds_loads + 2x WMMA | barrier.  Next-tile load latency hides behind WMMA.
// MODE 0: GI   -> f32 out = c + bias                     (TERN=false: plain bf16 convert)
// MODE 1: MLP1 -> bf16 out = relu(c + bias)              (TERN=true)
// MODE 2: MLP2 -> f32 out = c + bias + residual[m,n]     (TERN=true)
// MODE 3: HEAD -> f32 out = c + bias                     (TERN=true)
template <int MODE, bool TERN>
__global__ __launch_bounds__(256)
void gemm_wmma_k(const unsigned short* __restrict__ A, const float* __restrict__ W,
                 const float* __restrict__ scale, const float* __restrict__ bias,
                 const float* __restrict__ residual, void* __restrict__ Cout,
                 int M, int Nout, int K) {
    // Row stride 72 elems (144B = 36 DWORD banks): 9k mod 16 permutes the 4-bank
    // groups -> the 16 lanes of a half-wave hit 16 distinct bank groups.
    constexpr int LDA = 72;
    constexpr int LDB = 72;
    __shared__ unsigned short lA[32 * LDA];   // 4.5 KB
    __shared__ unsigned short lB[64 * LDB];   // 9.0 KB

    const int tid  = threadIdx.x;
    const int wave = tid >> 5;
    const int lane = tid & 31;
    const int half = lane >> 4;
    const int l16  = lane & 15;
    const int wm   = wave >> 2;            // 0..1 : M sub-tile
    const int wn   = wave & 3;             // 0..3 : N sub-tile
    const int m0   = blockIdx.y * 32;
    const int n0   = blockIdx.x * 64;

    float inv_s = 1.f;
    unsigned short bfP = 0, bfN = 0;
    if (TERN) {
        float s = scale[0];
        inv_s = 1.f / s;
        bfP = f2bf(s);
        bfN = f2bf(-s);
    }

    // staging coordinates (coalesced: 8 threads per A row, 16 per B row)
    const int ar = tid >> 3;               // A row 0..31
    const int ac = (tid & 7) * 8;          // A col 0,8,...,56  (us8 = 16B)
    const int br = tid >> 4;               // B row 0..15 (+16 per pass)
    const int bc = (tid & 15) * 4;         // B col 0,4,...,60  (f4 = 16B)

    const unsigned short* aGlob = A + (size_t)(m0 + ar) * K + ac;
    const float* wGlob0 = W + (size_t)(n0 + br)      * K + bc;
    const float* wGlob1 = W + (size_t)(n0 + br + 16) * K + bc;
    const float* wGlob2 = W + (size_t)(n0 + br + 32) * K + bc;
    const float* wGlob3 = W + (size_t)(n0 + br + 48) * K + bc;

    // prologue: load K-tile 0 into registers
    us8 av = *(const us8*)(aGlob);
    f4 wv0 = *(const f4*)(wGlob0);
    f4 wv1 = *(const f4*)(wGlob1);
    f4 wv2 = *(const f4*)(wGlob2);
    f4 wv3 = *(const f4*)(wGlob3);

    v8f c0 = {}, c1 = {};
    for (int k0 = 0; k0 < K; k0 += 64) {
        // ---- commit current register tile to LDS (quantize weights on the fly) ----
        *(us8*)(lA + ar * LDA + ac) = av;
        {
            f4 wv[4] = {wv0, wv1, wv2, wv3};
            #pragma unroll
            for (int p = 0; p < 4; ++p) {
                us4 qv;
                #pragma unroll
                for (int j = 0; j < 4; ++j) {
                    if (TERN) {
                        float r = wv[p][j] * inv_s;
                        qv[j] = (r >= 0.5f) ? bfP
                              : ((r <= -0.5f) ? bfN : (unsigned short)0);
                    } else {
                        qv[j] = f2bf(wv[p][j]);
                    }
                }
                *(us4*)(lB + (br + p * 16) * LDB + bc) = qv;
            }
        }
        __syncthreads();

        // ---- issue NEXT K-tile's global loads (consumed next iteration) ----
        const int kn = k0 + 64;
        if (kn < K) {
            av  = *(const us8*)(aGlob + kn);
            wv0 = *(const f4*)(wGlob0 + kn);
            wv1 = *(const f4*)(wGlob1 + kn);
            wv2 = *(const f4*)(wGlob2 + kn);
            wv3 = *(const f4*)(wGlob3 + kn);
            if (kn + 64 < K) __builtin_prefetch(wGlob0 + kn + 64, 0, 1);
        }

        // ---- fragments from LDS (16B ds_load_b128 each) + 2x WMMA ----
        FragU a0, a1, b0, b1;
        const unsigned short* aRow = lA + (wm * 16 + l16) * LDA;
        a0.h[0] = *(const us8*)(aRow +       8 * half);
        a0.h[1] = *(const us8*)(aRow + 16 +  8 * half);
        a1.h[0] = *(const us8*)(aRow + 32 +  8 * half);
        a1.h[1] = *(const us8*)(aRow + 48 +  8 * half);
        const unsigned short* bRow = lB + (wn * 16 + l16) * LDB;
        b0.h[0] = *(const us8*)(bRow +       8 * half);
        b0.h[1] = *(const us8*)(bRow + 16 +  8 * half);
        b1.h[0] = *(const us8*)(bRow + 32 +  8 * half);
        b1.h[1] = *(const us8*)(bRow + 48 +  8 * half);

        // two independent accumulators break the D->C serial chain
        c0 = __builtin_amdgcn_wmma_f32_16x16x32_bf16(false, a0.v, false, b0.v,
                                                     (short)0, c0, false, false);
        c1 = __builtin_amdgcn_wmma_f32_16x16x32_bf16(false, a1.v, false, b1.v,
                                                     (short)0, c1, false, false);
        __syncthreads();
    }

    // C layout: VGPR r, lanes 0-15 -> M=r; lanes 16-31 -> M=8+r; N = lane&15
    const int mw = m0 + wm * 16;
    const int n  = n0 + wn * 16 + l16;
    const float bn = bias[n];
    #pragma unroll
    for (int r = 0; r < 8; ++r) {
        int m = mw + r + 8 * half;
        float v = c0[r] + c1[r] + bn;
        if (MODE == 1) {
            v = fmaxf(v, 0.f);
            ((unsigned short*)Cout)[(size_t)m * Nout + n] = f2bf(v);
        } else {
            if (MODE == 2) v += residual[(size_t)m * Nout + n];
            ((float*)Cout)[(size_t)m * Nout + n] = v;
        }
    }
}

// ---------------- GRU recurrent step (fp32 VALU; M=2 is too small for WMMA) ----------------
__global__ __launch_bounds__(256) void gru_step_k(const float* __restrict__ gi,
                                                  const float* __restrict__ whh,
                                                  const float* __restrict__ bhh,
                                                  const float* __restrict__ h_in,
                                                  float* __restrict__ h_out,
                                                  float* __restrict__ xseq, int t) {
    __shared__ float hs[DD];
    int g = blockIdx.x * 256 + threadIdx.x;   // 0..B*D-1
    int b = g >> 10;
    int d = g & (DD - 1);
    const float* hb = h_in + (size_t)b * DD;
    for (int j = threadIdx.x; j < DD; j += 256) hs[j] = hb[j];
    __syncthreads();

    const f4* wr = (const f4*)(whh + (size_t)d * DD);
    const f4* wz = (const f4*)(whh + (size_t)(DD + d) * DD);
    const f4* wn = (const f4*)(whh + (size_t)(2 * DD + d) * DD);
    const f4* h4 = (const f4*)hs;
    float ar = 0.f, az = 0.f, an = 0.f;
    for (int k = 0; k < DD / 4; ++k) {
        f4 hv = h4[k];
        f4 r = wr[k], z = wz[k], nw = wn[k];
        #pragma unroll
        for (int j = 0; j < 4; ++j) {
            ar = fmaf(r[j],  hv[j], ar);
            az = fmaf(z[j],  hv[j], az);
            an = fmaf(nw[j], hv[j], an);
        }
    }
    int tok = b * SS + t;
    const float* gt = gi + (size_t)tok * D3;
    float rg = sigmoidf_(gt[d] + ar + bhh[d]);
    float zg = sigmoidf_(gt[DD + d] + az + bhh[DD + d]);
    float ng = tanhf(gt[2 * DD + d] + rg * (an + bhh[2 * DD + d]));
    float hn = (1.f - zg) * ng + zg * hs[d];
    h_out[(size_t)b * DD + d] = hn;
    xseq[(size_t)tok * DD + d] = hn;
}

// ---------------- LayerNorm (one block per token) ----------------
__global__ __launch_bounds__(256) void ln_k(const float* __restrict__ src,
                                            const float* __restrict__ g,
                                            const float* __restrict__ b,
                                            float* __restrict__ xout,
                                            unsigned short* __restrict__ xbf) {
    __shared__ float s1[256], s2[256];
    int tok = blockIdx.x;
    const float* r = src + (size_t)tok * DD;
    float v[4], sum = 0.f, sq = 0.f;
    #pragma unroll
    for (int j = 0; j < 4; ++j) {
        v[j] = r[threadIdx.x + j * 256];
        sum += v[j];
        sq  += v[j] * v[j];
    }
    s1[threadIdx.x] = sum;
    s2[threadIdx.x] = sq;
    __syncthreads();
    for (int s = 128; s > 0; s >>= 1) {
        if ((int)threadIdx.x < s) {
            s1[threadIdx.x] += s1[threadIdx.x + s];
            s2[threadIdx.x] += s2[threadIdx.x + s];
        }
        __syncthreads();
    }
    float mu  = s1[0] * (1.f / DD);
    float var = s2[0] * (1.f / DD) - mu * mu;
    float inv = rsqrtf(var + 1e-5f);
    #pragma unroll
    for (int j = 0; j < 4; ++j) {
        int d = threadIdx.x + j * 256;
        float y = (v[j] - mu) * inv * g[d] + b[d];
        xout[(size_t)tok * DD + d] = y;
        xbf[(size_t)tok * DD + d] = f2bf(y);
    }
}

// ---------------- small utility kernels ----------------
__global__ __launch_bounds__(256) void f2bf_k(const float* __restrict__ x,
                                              unsigned short* __restrict__ y, long n) {
    long i = (long)blockIdx.x * 256 + threadIdx.x;
    if (i < n) y[i] = f2bf(x[i]);
}

__global__ __launch_bounds__(256) void zero_k(float* __restrict__ p, long n) {
    long i = (long)blockIdx.x * 256 + threadIdx.x;
    if (i < n) p[i] = 0.f;
}

// ---------------- host launch ----------------
extern "C" void kernel_launch(void* const* d_in, const int* in_sizes, int n_in,
                              void* d_out, int out_size, void* d_ws, size_t ws_size,
                              hipStream_t stream) {
    (void)in_sizes; (void)n_in; (void)out_size; (void)ws_size;

    const int*   ids    = (const int*)  d_in[0];
    const float* emb    = (const float*)d_in[1];
    const float* pos    = (const float*)d_in[2];
    const float* wih    = (const float*)d_in[3];
    const float* whh    = (const float*)d_in[4];
    const float* bih    = (const float*)d_in[5];
    const float* bhh    = (const float*)d_in[6];
    const float* syn_w  = (const float*)d_in[7];
    const float* syn_b  = (const float*)d_in[8];
    const float* out_w  = (const float*)d_in[9];
    const float* out_b  = (const float*)d_in[10];
    const float* ln_g   = (const float*)d_in[11];
    const float* ln_b   = (const float*)d_in[12];
    const float* on_g   = (const float*)d_in[13];
    const float* on_b   = (const float*)d_in[14];
    const float* head_w = (const float*)d_in[15];
    const float* head_b = (const float*)d_in[16];
    float* logits = (float*)d_out;

    char* ws = (char*)d_ws;
    float*          scales = (float*)(ws + OFF_SCALES);
    float*          part   = (float*)(ws + OFF_PART);
    float*          X      = (float*)(ws + OFF_X);
    unsigned short* XBF    = (unsigned short*)(ws + OFF_XBF);
    float*          GI     = (float*)(ws + OFF_GI);
    float*          HA     = (float*)(ws + OFF_HA);
    float*          HB     = (float*)(ws + OFF_HB);
    unsigned short* HID    = (unsigned short*)(ws + OFF_HID);
    float*          PRELN  = (float*)(ws + OFF_PRELN);

    // 1) per-tensor abs-mean scales (deterministic two-stage reduction)
    auto reduce = [&](const float* w, long n, int idx) {
        absmean_part_k<<<256, 256, 0, stream>>>(w, n, part);
        absmean_fin_k<<<1, 256, 0, stream>>>(part, scales, idx, 1.f / (float)n);
    };
    reduce(emb, (long)VV * DD, 0);
    for (int l = 0; l < LL; ++l) reduce(syn_w + (size_t)l * NN * DD, (long)NN * DD, 1 + l);
    for (int l = 0; l < LL; ++l) reduce(out_w + (size_t)l * DD * NN, (long)DD * NN, 5 + l);
    reduce(head_w, (long)VV * DD, 9);

    // 2) ternary embedding lookup + positional add
    embed_k<<<(TT * DD) / 256, 256, 0, stream>>>(ids, emb, pos, scales, X, XBF);

    // 3) GRU input projection for all timesteps: GI = X * wih^T + bih  (WMMA, bf16)
    gemm_wmma_k<0, false><<<dim3(D3 / 64, TT / 32), 256, 0, stream>>>(
        XBF, wih, nullptr, bih, nullptr, GI, TT, D3, DD);

    // 4) GRU recurrence: 512 sequential steps (fp32 VALU; latency-bound, M=2)
    zero_k<<<(BB * DD + 255) / 256, 256, 0, stream>>>(HA, (long)BB * DD);
    for (int t = 0; t < SS; ++t) {
        const float* hin  = (t & 1) ? HB : HA;
        float*       hout = (t & 1) ? HA : HB;
        gru_step_k<<<(BB * DD) / 256, 256, 0, stream>>>(GI, whh, bhh, hin, hout, X, t);
    }
    f2bf_k<<<(TT * DD) / 256, 256, 0, stream>>>(X, XBF, (long)TT * DD);

    // 5) ternary MLP blocks (WMMA GEMMs with fused epilogues)
    for (int l = 0; l < LL; ++l) {
        gemm_wmma_k<1, true><<<dim3(NN / 64, TT / 32), 256, 0, stream>>>(
            XBF, syn_w + (size_t)l * NN * DD, scales + 1 + l, syn_b + (size_t)l * NN,
            nullptr, HID, TT, NN, DD);
        gemm_wmma_k<2, true><<<dim3(DD / 64, TT / 32), 256, 0, stream>>>(
            HID, out_w + (size_t)l * DD * NN, scales + 5 + l, out_b + (size_t)l * DD,
            X, PRELN, TT, DD, NN);
        ln_k<<<TT, 256, 0, stream>>>(PRELN, ln_g + (size_t)l * DD, ln_b + (size_t)l * DD,
                                     X, XBF);
    }

    // 6) final LayerNorm (in-place safe: values cached in registers)
    ln_k<<<TT, 256, 0, stream>>>(X, on_g, on_b, X, XBF);

    // 7) ternary head: logits = X * head_w^T + head_b  (WMMA)
    gemm_wmma_k<3, true><<<dim3(VV / 64, TT / 32), 256, 0, stream>>>(
        XBF, head_w, scales + 9, head_b, nullptr, logits, TT, VV, DD);
}
